// LSTMTagger_37099927503213
// MI455X (gfx1250) — matrix operand order repaired and checked by reading, not compiled, not run on hardware
//
#include <hip/hip_runtime.h>

// ---------------------------------------------------------------------------
// Types for CDNA5 WMMA f16 (16x16x32, f32 accum)
// ---------------------------------------------------------------------------
typedef _Float16 half_t;
typedef __attribute__((ext_vector_type(16))) _Float16 v16h;
typedef __attribute__((ext_vector_type(8)))  _Float16 v8h;
typedef __attribute__((ext_vector_type(8)))  float    v8f;
typedef __attribute__((ext_vector_type(4)))  unsigned int v4u;
typedef __attribute__((ext_vector_type(8)))  int          v8i;
typedef __attribute__((ext_vector_type(4)))  int          v4i;

#define LSTM_B 64
#define LSTM_S 512
#define LSTM_H 1024

#if __has_builtin(__builtin_amdgcn_tensor_load_to_lds)
#define HAVE_TDM 1
#else
#define HAVE_TDM 0
#endif

// A-fragment (16x32, MxK), row-major source, lda in elements.
// Lane L: M = L&15; halves split K: half0 -> K in [0..15], half1 -> [16..31].
__device__ __forceinline__ v16h load_a_frag(const half_t* tile, size_t lda, int lane) {
    const int m  = lane & 15;
    const int kh = (lane >> 4) * 16;
    const half_t* p = tile + (size_t)m * lda + kh;
    v8h lo = *reinterpret_cast<const v8h*>(p);        // K = kh .. kh+7
    v8h hi = *reinterpret_cast<const v8h*>(p + 8);    // K = kh+8 .. kh+15
    v16h r;
#pragma unroll
    for (int e = 0; e < 8; ++e) { r[e] = lo[e]; r[e + 8] = hi[e]; }
    return r;
}

// B-fragment (32x16, KxN) sourced from W stored (N,K) row-major: B[k][n] = W[n][k].
// Lane L: N = L&15, K = (L>>4)*16 + e  -> one contiguous 32-byte read per lane.
__device__ __forceinline__ v16h load_b_frag(const half_t* wtile, size_t ldw, int lane) {
    const int n = lane & 15;
    const int k = (lane >> 4) * 16;
    return *reinterpret_cast<const v16h*>(wtile + (size_t)n * ldw + k);
}

__device__ __forceinline__ v8f wmma_f16(v16h a, v16h b, v8f c) {
    return __builtin_amdgcn_wmma_f32_16x16x32_f16(
        /*neg_a=*/false, a, /*neg_b=*/false, b,
        /*c_mod=*/(short)0, c, /*reuse_a=*/false, /*reuse_b=*/false);
}

// ---------------------------------------------------------------------------
// TDM: stage a 2D f16 tile (rows x cols, row stride in elements) into LDS.
// D# built per CDNA5 ISA §8.3/8.4. Issued by one wave; EXEC is ignored by TDM.
// ---------------------------------------------------------------------------
__device__ __forceinline__ void tdm_load_tile_f16(unsigned lds_off, const half_t* gptr,
                                                  unsigned rows, unsigned cols,
                                                  unsigned row_stride_elems) {
#if HAVE_TDM
    const unsigned long long ga = (unsigned long long)(size_t)gptr;
    // Group 0: count=1 | lds_addr | global_addr[56:0] | type=2
    v4u g0 = { 1u,
               lds_off,
               (unsigned)(ga & 0xFFFFFFFFu),
               ((unsigned)((ga >> 32) & 0x01FFFFFFu)) | (2u << 30) };
    // Group 1: wg_mask=0, data_size=1 (2B), no barrier/iterate/pad,
    //          tensor_dim0=cols, tensor_dim1=rows, tile_dim0=cols, tile_dim1=rows,
    //          tensor_dim0_stride=row_stride_elems.
    v8i g1 = { (int)(1u << 16),                              // [17:16] data_size = 2 bytes
               (int)(cols << 16),                            // [63:48] tensor_dim0 lo16
               (int)(((cols >> 16) & 0xFFFFu) | (rows << 16)),// [79:64] dim0 hi | [95:80] dim1 lo
               (int)(cols << 16),                            // [111:96] dim1 hi(0) | [127:112] tile_dim0
               (int)rows,                                    // [143:128] tile_dim1 | [159:144] tile_dim2=0
               (int)row_stride_elems,                        // [207:160] tensor_dim0_stride lo32
               0,                                            // stride hi | dim1_stride lo (unused, 2D)
               0 };
    v4i g2 = { 0, 0, 0, 0 };   // 2D tile: groups 2/3 unused
    v4i g3 = { 0, 0, 0, 0 };
#if defined(__clang_major__) && (__clang_major__ >= 23)
    v8i gx = { 0, 0, 0, 0, 0, 0, 0, 0 };
    __builtin_amdgcn_tensor_load_to_lds(g0, g1, g2, g3, gx, 0);
#else
    __builtin_amdgcn_tensor_load_to_lds(g0, g1, g2, g3, 0);
#endif
#else
    (void)lds_off; (void)gptr; (void)rows; (void)cols; (void)row_stride_elems;
#endif
}

// ---------------------------------------------------------------------------
// Generic GEMM: C[m,n] = sum_k A[m,k] * W[n,k] (+ bias[n]).
// A (M,K) f16 row-major, W (N,K) f16 row-major, C (M,N) f32 row-major.
// ---------------------------------------------------------------------------
template<int WAVES_M, int WAVES_N, int FRAG_M, int FRAG_N>
__global__ __launch_bounds__(WAVES_M * WAVES_N * 32)
void gemm_f16_wmma(const half_t* __restrict__ A, const half_t* __restrict__ W,
                   const float* __restrict__ bias, float* __restrict__ C,
                   int K, size_t lda, size_t ldw, size_t ldc) {
    const int lane = threadIdx.x & 31;
    const int wave = threadIdx.x >> 5;
    const int wm   = wave / WAVES_N;
    const int wn   = wave % WAVES_N;
    const int BM   = WAVES_M * FRAG_M * 16;
    const int BN   = WAVES_N * FRAG_N * 16;
    const size_t m0 = (size_t)blockIdx.y * BM + (size_t)wm * FRAG_M * 16;
    const size_t n0 = (size_t)blockIdx.x * BN + (size_t)wn * FRAG_N * 16;

    v8f acc[FRAG_M][FRAG_N];
#pragma unroll
    for (int i = 0; i < FRAG_M; ++i)
#pragma unroll
        for (int j = 0; j < FRAG_N; ++j)
#pragma unroll
            for (int v = 0; v < 8; ++v) acc[i][j][v] = 0.f;

    for (int k0 = 0; k0 < K; k0 += 32) {
        v16h a[FRAG_M], b[FRAG_N];
#pragma unroll
        for (int i = 0; i < FRAG_M; ++i)
            a[i] = load_a_frag(A + (m0 + (size_t)i * 16) * lda + k0, lda, lane);
#pragma unroll
        for (int j = 0; j < FRAG_N; ++j)
            b[j] = load_b_frag(W + (n0 + (size_t)j * 16) * ldw + k0, ldw, lane);

        if (k0 + 32 < K) {  // hint next K-slab (global_prefetch_b8)
            __builtin_prefetch(A + (m0 + (size_t)(lane & (FRAG_M * 16 - 1))) * lda + k0 + 32, 0, 3);
            __builtin_prefetch(W + (n0 + (size_t)(lane & (FRAG_N * 16 - 1))) * ldw + k0 + 32, 0, 3);
        }
#pragma unroll
        for (int i = 0; i < FRAG_M; ++i)
#pragma unroll
            for (int j = 0; j < FRAG_N; ++j)
                acc[i][j] = wmma_f16(a[i], b[j], acc[i][j]);
    }

    const int hh = lane >> 4, nl = lane & 15;  // C layout: lane=col, vgpr v -> row v + 8*hh
#pragma unroll
    for (int i = 0; i < FRAG_M; ++i)
#pragma unroll
        for (int j = 0; j < FRAG_N; ++j) {
            const size_t col = n0 + (size_t)j * 16 + nl;
            const float bv = bias ? bias[col] : 0.f;
#pragma unroll
            for (int v = 0; v < 8; ++v) {
                const size_t row = m0 + (size_t)i * 16 + v + 8 * hh;
                C[row * ldc + col] = acc[i][j][v] + bv;
            }
        }
}

// ---------------------------------------------------------------------------
// Device-wide split barrier (arrive + spin) for the persistent LSTM kernel.
// ---------------------------------------------------------------------------
__device__ __forceinline__ void grid_barrier(unsigned* cnt, unsigned* gen, unsigned nblocks) {
    __threadfence();          // flush this thread's h-stores to device scope
    __syncthreads();
    if (threadIdx.x == 0) {
        const unsigned g = __hip_atomic_load(gen, __ATOMIC_RELAXED, __HIP_MEMORY_SCOPE_AGENT);
        const unsigned a = __hip_atomic_fetch_add(cnt, 1u, __ATOMIC_ACQ_REL, __HIP_MEMORY_SCOPE_AGENT);
        if (a == nblocks - 1u) {
            __hip_atomic_store(cnt, 0u, __ATOMIC_RELAXED, __HIP_MEMORY_SCOPE_AGENT);
            __hip_atomic_store(gen, g + 1u, __ATOMIC_RELEASE, __HIP_MEMORY_SCOPE_AGENT);
        } else {
            while (__hip_atomic_load(gen, __ATOMIC_ACQUIRE, __HIP_MEMORY_SCOPE_AGENT) == g)
                __builtin_amdgcn_s_sleep(2);
        }
    }
    __syncthreads();
}

// ---------------------------------------------------------------------------
// Persistent LSTM layer. 64 blocks; block owns hidden units [j0, j0+16).
// One-time: TDM-stage the block's Whh slice (4 gates x 16 rows x 1024 f16 =
// 128 KB) into LDS. Then for each of 512 timesteps: WMMA gate GEMM with
// A-fragments of h_{t-1} from global and B-fragments from LDS, gate exchange
// through LDS, c update in registers, h written f16 to hseq, grid barrier.
// LDS: 128 KB weights + 16 KB gate buffer = 144 KB -> 2 blocks/WGP (of 320 KB),
// so all 64 blocks are co-resident and the barrier cannot deadlock.
// ---------------------------------------------------------------------------
__global__ __launch_bounds__(256)
void lstm_layer_persistent(const float* __restrict__ xproj,   // (B*S, 4H), row=b*S+t, biases folded in
                           const half_t* __restrict__ Whh,    // (4H, H) f16 row-major
                           half_t* __restrict__ hseq,         // h at (b*S+t)*H + j
                           unsigned* __restrict__ bar) {      // [0]=count, [1]=generation (zeroed)
    extern __shared__ char smem[];
    half_t* lds_w = (half_t*)smem;                                   // 4*16*1024 f16
    float*  gbuf  = (float*)(smem + 4 * 16 * LSTM_H * sizeof(half_t)); // 4*64*16 f32

    const int lane = threadIdx.x & 31;
    const int wave = threadIdx.x >> 5;
    const int gate = wave >> 1;          // 0:i 1:f 2:g 3:o
    const int mh   = wave & 1;           // batch rows mh*32 .. mh*32+31
    const int j0   = blockIdx.x * 16;
    const int hh   = lane >> 4, nl = lane & 15;

    // ---- one-time weight staging into LDS ----
#if HAVE_TDM
    if (mh == 0) {  // 4 waves issue one TDM descriptor each (wave-uniform branch)
        const unsigned lds_off = (unsigned)(size_t)(lds_w + (size_t)gate * 16 * LSTM_H);
        tdm_load_tile_f16(lds_off, Whh + ((size_t)gate * LSTM_H + j0) * LSTM_H,
                          16u, (unsigned)LSTM_H, (unsigned)LSTM_H);
        __builtin_amdgcn_s_wait_tensorcnt(0);
    }
#else
    for (int idx = threadIdx.x; idx < 64 * (LSTM_H / 8); idx += 256) {
        const int row = idx / (LSTM_H / 8);
        const int kc  = (idx % (LSTM_H / 8)) * 8;
        const int g   = row >> 4, r = row & 15;
        *reinterpret_cast<v8h*>(lds_w + (size_t)row * LSTM_H + kc) =
            *reinterpret_cast<const v8h*>(Whh + ((size_t)(g * LSTM_H + j0 + r)) * LSTM_H + kc);
    }
#endif
    __syncthreads();

    const size_t lda = (size_t)LSTM_S * LSTM_H;           // batch-row stride in hseq
    const half_t* ldsB = lds_w + (size_t)gate * 16 * LSTM_H;

    float cstate[4] = {0.f, 0.f, 0.f, 0.f};               // this thread's fixed (b,j) cells

    for (int t = 0; t < LSTM_S; ++t) {
        v8f acc[2];
#pragma unroll
        for (int v = 0; v < 8; ++v) { acc[0][v] = 0.f; acc[1][v] = 0.f; }

        if (t > 0) {  // uniform branch: EXEC stays all-1s around WMMA
            const half_t* hprev = hseq + (size_t)(t - 1) * LSTM_H;
            for (int k0 = 0; k0 < LSTM_H; k0 += 32) {
                v16h a0 = load_a_frag(hprev + (size_t)(mh * 32)      * lda + k0, lda, lane);
                v16h a1 = load_a_frag(hprev + (size_t)(mh * 32 + 16) * lda + k0, lda, lane);
                v16h b  = load_b_frag(ldsB + k0, (size_t)LSTM_H, lane);  // ds_load from LDS
                acc[0] = wmma_f16(a0, b, acc[0]);
                acc[1] = wmma_f16(a1, b, acc[1]);
            }
        }

#pragma unroll
        for (int i = 0; i < 2; ++i)
#pragma unroll
            for (int v = 0; v < 8; ++v)
                gbuf[((gate * 64) + mh * 32 + i * 16 + v + 8 * hh) * 16 + nl] = acc[i][v];
        __syncthreads();

        // elementwise: thread owns 4 fixed (b, jl) cells across all timesteps
#pragma unroll
        for (int r = 0; r < 4; ++r) {
            const int e  = threadIdx.x + 256 * r;
            const int b  = e >> 4;
            const int jl = e & 15;
            const int j  = j0 + jl;
            const float* xp = xproj + ((size_t)b * LSTM_S + t) * (4 * LSTM_H);
            float gi = gbuf[(0 * 64 + b) * 16 + jl] + xp[0 * LSTM_H + j];
            float gf = gbuf[(1 * 64 + b) * 16 + jl] + xp[1 * LSTM_H + j];
            float gg = gbuf[(2 * 64 + b) * 16 + jl] + xp[2 * LSTM_H + j];
            float go = gbuf[(3 * 64 + b) * 16 + jl] + xp[3 * LSTM_H + j];
            gi = 1.f / (1.f + __expf(-gi));
            gf = 1.f / (1.f + __expf(-gf));
            gg = tanhf(gg);
            go = 1.f / (1.f + __expf(-go));
            cstate[r] = gf * cstate[r] + gi * gg;
            hseq[((size_t)b * LSTM_S + t) * LSTM_H + j] = (half_t)(go * tanhf(cstate[r]));
        }

        if (t + 1 < LSTM_S)
            grid_barrier(bar, bar + 1, gridDim.x);  // also orders gbuf reuse
    }
}

// ---------------------------------------------------------------------------
// Helpers: f32 -> f16 conversion, bias sum
// ---------------------------------------------------------------------------
__global__ void cvt_f32_to_f16(const float* __restrict__ src, half_t* __restrict__ dst, size_t n) {
    size_t i = (size_t)blockIdx.x * blockDim.x + threadIdx.x;
    const size_t stride = (size_t)gridDim.x * blockDim.x;
    for (; i < n; i += stride) dst[i] = (half_t)src[i];
}

__global__ void bias_sum(const float* __restrict__ a, const float* __restrict__ b,
                         float* __restrict__ o, int n) {
    const int i = blockIdx.x * blockDim.x + threadIdx.x;
    if (i < n) o[i] = a[i] + b[i];
}

// ---------------------------------------------------------------------------
// Host orchestration
// ---------------------------------------------------------------------------
extern "C" void kernel_launch(void* const* d_in, const int* in_sizes, int n_in,
                              void* d_out, int out_size, void* d_ws, size_t ws_size,
                              hipStream_t stream) {
    (void)in_sizes; (void)n_in; (void)out_size; (void)ws_size;
    const int B = LSTM_B, S = LSTM_S, I = 512, H = LSTM_H, O = 64;

    const float* x    = (const float*)d_in[0];
    const float* Wih0 = (const float*)d_in[1];
    const float* Whh0 = (const float*)d_in[2];
    const float* bih0 = (const float*)d_in[3];
    const float* bhh0 = (const float*)d_in[4];
    const float* Wih1 = (const float*)d_in[5];
    const float* Whh1 = (const float*)d_in[6];
    const float* bih1 = (const float*)d_in[7];
    const float* bhh1 = (const float*)d_in[8];
    const float* Wfc  = (const float*)d_in[9];
    const float* bfc  = (const float*)d_in[10];

    char* ws = (char*)d_ws;
    size_t off = 0;
    auto take = [&](size_t bytes) -> char* {
        char* p = ws + off;
        off += (bytes + 255) & ~(size_t)255;
        return p;
    };
    half_t*   x_h    = (half_t*)take((size_t)B * S * I * sizeof(half_t));
    half_t*   Wih0_h = (half_t*)take((size_t)4 * H * I * sizeof(half_t));
    half_t*   Whh0_h = (half_t*)take((size_t)4 * H * H * sizeof(half_t));
    half_t*   Wih1_h = (half_t*)take((size_t)4 * H * H * sizeof(half_t));
    half_t*   Whh1_h = (half_t*)take((size_t)4 * H * H * sizeof(half_t));
    half_t*   Wfc_h  = (half_t*)take((size_t)O * H * sizeof(half_t));
    float*    bsum0  = (float*)take((size_t)4 * H * sizeof(float));
    float*    bsum1  = (float*)take((size_t)4 * H * sizeof(float));
    float*    xproj  = (float*)take((size_t)B * S * 4 * H * sizeof(float));  // reused per layer
    half_t*   hseq0  = (half_t*)take((size_t)B * S * H * sizeof(half_t));
    half_t*   hseq1  = (half_t*)take((size_t)B * S * H * sizeof(half_t));
    unsigned* bar    = (unsigned*)take(256);

    auto cvt = [&](const float* s, half_t* d, size_t n) {
        int blocks = (int)((n + 256 * 8 - 1) / (256 * 8));
        if (blocks > 4096) blocks = 4096;
        cvt_f32_to_f16<<<blocks, 256, 0, stream>>>(s, d, n);
    };
    cvt(x,    x_h,    (size_t)B * S * I);
    cvt(Wih0, Wih0_h, (size_t)4 * H * I);
    cvt(Whh0, Whh0_h, (size_t)4 * H * H);
    cvt(Wih1, Wih1_h, (size_t)4 * H * H);
    cvt(Whh1, Whh1_h, (size_t)4 * H * H);
    cvt(Wfc,  Wfc_h,  (size_t)O * H);
    bias_sum<<<(4 * H + 255) / 256, 256, 0, stream>>>(bih0, bhh0, bsum0, 4 * H);
    bias_sum<<<(4 * H + 255) / 256, 256, 0, stream>>>(bih1, bhh1, bsum1, 4 * H);

    const size_t lstm_lds = (size_t)4 * 16 * H * sizeof(half_t) + (size_t)4 * 64 * 16 * sizeof(float);

    // ---- Layer 0 ----
    {   // xproj = x @ Wih0^T + (bih0+bhh0) : M=B*S, N=4H, K=I, tile 128x128
        dim3 grid((4 * H) / 128, (B * S) / 128);
        gemm_f16_wmma<2, 4, 4, 2><<<grid, 256, 0, stream>>>(
            x_h, Wih0_h, bsum0, xproj, I, (size_t)I, (size_t)I, (size_t)4 * H);
    }
    hipMemsetAsync(bar, 0, 2 * sizeof(unsigned), stream);
    lstm_layer_persistent<<<H / 16, 256, lstm_lds, stream>>>(xproj, Whh0_h, hseq0, bar);

    // ---- Layer 1 ----
    {   // xproj = hseq0 @ Wih1^T + (bih1+bhh1) : M=B*S, N=4H, K=H
        dim3 grid((4 * H) / 128, (B * S) / 128);
        gemm_f16_wmma<2, 4, 4, 2><<<grid, 256, 0, stream>>>(
            hseq0, Wih1_h, bsum1, xproj, H, (size_t)H, (size_t)H, (size_t)4 * H);
    }
    hipMemsetAsync(bar, 0, 2 * sizeof(unsigned), stream);
    lstm_layer_persistent<<<H / 16, 256, lstm_lds, stream>>>(xproj, Whh1_h, hseq1, bar);

    // ---- FC: logits = hseq1 @ Wfc^T + bfc : M=B*S, N=64, K=H, tile 128x64 ----
    {
        dim3 grid(O / 64, (B * S) / 128);
        gemm_f16_wmma<4, 2, 2, 2><<<grid, 256, 0, stream>>>(
            hseq1, Wfc_h, bfc, (float*)d_out, H, (size_t)H, (size_t)H, (size_t)O);
    }
}
